// GatedAttentionLayer_16552803959278
// MI455X (gfx1250) — matrix operand order repaired and checked by reading, not compiled, not run on hardware
//
#include <hip/hip_runtime.h>

typedef __attribute__((ext_vector_type(16))) _Float16 v16h;
typedef __attribute__((ext_vector_type(8)))  _Float16 v8h;
typedef __attribute__((ext_vector_type(4)))  _Float16 v4h;
typedef __attribute__((ext_vector_type(8)))  float    v8f;

#define NB   8      // batch
#define LQ   512    // query length
#define LC   2048   // context length
#define DD   768    // embedding dim
#define KC   64     // K chunk staged per barrier (2 WMMA sub-steps of 32)
#define LDP  72     // padded LDS row stride (f16) -> 144B rows: 16B aligned, conflict-free

// ---- WMMA fragment loaders (16-bit 16x16x32 layouts, CDNA5 ISA 7.12.2) ----
// A (16x32 MxK): lane L -> row M=L&15; K = {h..h+7} U {16+h..16+h+7}, h = (L&16)?8:0
__device__ __forceinline__ v16h frag_a(const _Float16* lds, int row, int lane, int ks) {
  const _Float16* p = lds + row * LDP + ks + ((lane & 16) ? 8 : 0);
  v8h lo = *(const v8h*)(p);
  v8h hi = *(const v8h*)(p + 16);
  v16h r;
#pragma unroll
  for (int i = 0; i < 8; ++i) { r[i] = lo[i]; r[i + 8] = hi[i]; }
  return r;
}
// B (32x16 KxN) staged [n][k]: lane L -> col N=L&15; K = {h..h+7} U {16+h..16+h+7}, h = (L&16)?8:0
__device__ __forceinline__ v16h frag_b(const _Float16* lds, int col, int lane, int ks) {
  const _Float16* p = lds + col * LDP + ks + ((lane & 16) ? 8 : 0);
  v8h lo = *(const v8h*)(p);
  v8h hi = *(const v8h*)(p + 16);
  v16h r;
#pragma unroll
  for (int i = 0; i < 8; ++i) { r[i] = lo[i]; r[i + 8] = hi[i]; }
  return r;
}

__device__ __forceinline__ v4h cvt4(float x, float y, float z, float w) {
  v4h h;
  h[0] = (_Float16)x; h[1] = (_Float16)y; h[2] = (_Float16)z; h[3] = (_Float16)w;
  return h;
}

// ============ Kernel 1: S[b,q,c] = sum_d Q[b,q,d] * C[b,c,d] ============
// Block tile 64 (q) x 128 (c), 8 waves: wm = wave&3 (M strip), wn = wave>>2 (N half)
__global__ __launch_bounds__(256) void gemm1_qk(const float* __restrict__ qry,
                                                const float* __restrict__ ctx,
                                                float* __restrict__ S) {
  __shared__ _Float16 lA[64 * LDP];
  __shared__ _Float16 lB[128 * LDP];
  const int b  = blockIdx.z;
  const int q0 = blockIdx.y * 64;
  const int c0 = blockIdx.x * 128;
  const int t = threadIdx.x, wave = t >> 5, lane = t & 31;
  const int wm = wave & 3, wn = wave >> 2;

  const float* qb = qry + (size_t)b * LQ * DD;
  const float* cb = ctx + (size_t)b * LC * DD;

  v8f acc[4] = {};
  for (int k0 = 0; k0 < DD; k0 += KC) {
    // stage A: 64 rows (q) x 64 k : float4 loads -> packed v4h LDS stores
#pragma unroll
    for (int i = 0; i < (64 * KC) / (4 * 256); ++i) {
      int u = i * 256 + t;
      int r = u >> 4, k4 = (u & 15) * 4;
      float4 v = *(const float4*)(qb + (size_t)(q0 + r) * DD + k0 + k4);
      *(v4h*)(lA + r * LDP + k4) = cvt4(v.x, v.y, v.z, v.w);
    }
    // stage B: 128 rows (c) x 64 k
#pragma unroll
    for (int i = 0; i < (128 * KC) / (4 * 256); ++i) {
      int u = i * 256 + t;
      int r = u >> 4, k4 = (u & 15) * 4;
      float4 v = *(const float4*)(cb + (size_t)(c0 + r) * DD + k0 + k4);
      *(v4h*)(lB + r * LDP + k4) = cvt4(v.x, v.y, v.z, v.w);
    }
    __syncthreads();
#pragma unroll
    for (int ks = 0; ks < KC; ks += 32) {
      v16h a = frag_a(lA, wm * 16 + (lane & 15), lane, ks);
#pragma unroll
      for (int s = 0; s < 4; ++s) {
        v16h bf = frag_b(lB, wn * 64 + s * 16 + (lane & 15), lane, ks);
        acc[s] = __builtin_amdgcn_wmma_f32_16x16x32_f16(false, a, false, bf,
                                                        (short)0, acc[s], false, false);
      }
    }
    __syncthreads();
  }
  // epilogue: C/D layout -> M = r + 8*(lane>=16), N = lane&15
  const int mhi = (lane & 16) ? 8 : 0;
#pragma unroll
  for (int s = 0; s < 4; ++s) {
    int cg = c0 + wn * 64 + s * 16 + (lane & 15);
#pragma unroll
    for (int r = 0; r < 8; ++r) {
      int qg = q0 + wm * 16 + r + mhi;
      S[((size_t)b * LQ + qg) * LC + cg] = acc[s][r];
    }
  }
}

// ============ Kernel 2: per-(b,c) max and 1/sum of exp over q (x4 vectorized) ============
__global__ __launch_bounds__(256) void col_softmax_stats(const float* __restrict__ S,
                                                         float* __restrict__ cmax,
                                                         float* __restrict__ cinv) {
  const int b  = blockIdx.y;
  const int c4 = (blockIdx.x * 256 + threadIdx.x) * 4;
  const float* Sb = S + (size_t)b * LQ * LC;
  float4 m = make_float4(-3.4e38f, -3.4e38f, -3.4e38f, -3.4e38f);
  for (int qi = 0; qi < LQ; ++qi) {
    float4 v = *(const float4*)(Sb + (size_t)qi * LC + c4);
    m.x = fmaxf(m.x, v.x); m.y = fmaxf(m.y, v.y);
    m.z = fmaxf(m.z, v.z); m.w = fmaxf(m.w, v.w);
  }
  float4 s = make_float4(0.f, 0.f, 0.f, 0.f);
  for (int qi = 0; qi < LQ; ++qi) {
    float4 v = *(const float4*)(Sb + (size_t)qi * LC + c4);
    s.x += __expf(v.x - m.x); s.y += __expf(v.y - m.y);
    s.z += __expf(v.z - m.z); s.w += __expf(v.w - m.w);
  }
  *(float4*)(cmax + b * LC + c4) = m;
  *(float4*)(cinv + b * LC + c4) = make_float4(1.f / s.x, 1.f / s.y, 1.f / s.z, 1.f / s.w);
}

// ===== Kernel 3: out[b,c,d] = ctx[b,c,d] * sum_q alpha[q,c] * Q[q,d] =====
// A = alpha^T (M=c, K=q), B = Q^T-staged (K=q, N=d). Block tile 64 (c) x 128 (d).
__global__ __launch_bounds__(256) void gemm2_gate(const float* __restrict__ qry,
                                                  const float* __restrict__ ctx,
                                                  const float* __restrict__ S,
                                                  const float* __restrict__ cmax,
                                                  const float* __restrict__ cinv,
                                                  float* __restrict__ out) {
  __shared__ _Float16 lA[64 * LDP];
  __shared__ _Float16 lB[128 * LDP];
  __shared__ float sMax[64];
  __shared__ float sInv[64];
  const int b  = blockIdx.z;
  const int c0 = blockIdx.y * 64;
  const int d0 = blockIdx.x * 128;
  const int t = threadIdx.x, wave = t >> 5, lane = t & 31;
  const int wm = wave & 3, wn = wave >> 2;

  if (t < 64) {
    sMax[t] = cmax[b * LC + c0 + t];
    sInv[t] = cinv[b * LC + c0 + t];
  }
  __syncthreads();

  const float* Sb = S + (size_t)b * LQ * LC;
  const float* qb = qry + (size_t)b * LQ * DD;

  v8f acc[4] = {};
  for (int k0 = 0; k0 < LQ; k0 += KC) {
    // stage A: alpha^T tile, lA[m][k] = exp(S[k][m]-max)*inv; each thread: fixed m,
    // 4 consecutive k (adjacent lanes -> adjacent c: coalesced); one b64 LDS store
#pragma unroll
    for (int i = 0; i < (64 * KC) / (4 * 256); ++i) {
      int u = i * 256 + t;
      int m  = u & 63;
      int k4 = (u >> 6) * 4;
      float mx = sMax[m], iv = sInv[m];
      const float* sp = Sb + (size_t)(k0 + k4) * LC + (c0 + m);
      float a0 = __expf(sp[0]        - mx) * iv;
      float a1 = __expf(sp[LC]       - mx) * iv;
      float a2 = __expf(sp[2 * LC]   - mx) * iv;
      float a3 = __expf(sp[3 * LC]   - mx) * iv;
      *(v4h*)(lA + m * LDP + k4) = cvt4(a0, a1, a2, a3);
    }
    // stage B: lB[n][k] = Q[k][n]; fixed n, 4 consecutive k; one b64 LDS store
#pragma unroll
    for (int i = 0; i < (128 * KC) / (4 * 256); ++i) {
      int u = i * 256 + t;
      int n  = u & 127;
      int k4 = (u >> 7) * 4;
      const float* qp = qb + (size_t)(k0 + k4) * DD + (d0 + n);
      *(v4h*)(lB + n * LDP + k4) = cvt4(qp[0], qp[DD], qp[2 * DD], qp[3 * DD]);
    }
    __syncthreads();
#pragma unroll
    for (int ks = 0; ks < KC; ks += 32) {
      v16h a = frag_a(lA, wm * 16 + (lane & 15), lane, ks);
#pragma unroll
      for (int s = 0; s < 4; ++s) {
        v16h bf = frag_b(lB, wn * 64 + s * 16 + (lane & 15), lane, ks);
        acc[s] = __builtin_amdgcn_wmma_f32_16x16x32_f16(false, a, false, bf,
                                                        (short)0, acc[s], false, false);
      }
    }
    __syncthreads();
  }
  // fused gate epilogue
  const int mhi = (lane & 16) ? 8 : 0;
#pragma unroll
  for (int s = 0; s < 4; ++s) {
    int dg = d0 + wn * 64 + s * 16 + (lane & 15);
#pragma unroll
    for (int r = 0; r < 8; ++r) {
      int cg = c0 + wm * 16 + r + mhi;
      size_t off = ((size_t)b * LC + cg) * DD + dg;
      out[off] = acc[s][r] * ctx[off];
    }
  }
}

extern "C" void kernel_launch(void* const* d_in, const int* in_sizes, int n_in,
                              void* d_out, int out_size, void* d_ws, size_t ws_size,
                              hipStream_t stream) {
  const float* ctx = (const float*)d_in[0];  // [8, 2048, 768]
  const float* qry = (const float*)d_in[1];  // [8, 512, 768]
  float* out = (float*)d_out;                // [8, 2048, 768]

  // workspace: S (33.5 MB, L2-resident) + colMax + colInv
  float* S    = (float*)d_ws;
  float* cmax = S + (size_t)NB * LQ * LC;
  float* cinv = cmax + (size_t)NB * LC;

  dim3 g1(LC / 128, LQ / 64, NB);   // (16, 8, 8)
  gemm1_qk<<<g1, 256, 0, stream>>>(qry, ctx, S);

  dim3 g2(LC / 1024, NB);           // (2, 8)
  col_softmax_stats<<<g2, 256, 0, stream>>>(S, cmax, cinv);

  dim3 g3(DD / 128, LC / 64, NB);   // (6, 32, 8)
  gemm2_gate<<<g3, 256, 0, stream>>>(qry, ctx, S, cmax, cinv, out);
}
